// UniversalGeometricNetwork_37898791420146
// MI455X (gfx1250) — matrix operand order repaired and checked by reading, not compiled, not run on hardware
//
#include <hip/hip_runtime.h>
#include <math.h>

// ---------------------------------------------------------------------------
// Shapes (fixed by the reference)
// ---------------------------------------------------------------------------
#define BATCH 65536
#define DIN   1024
#define DOUT  1024
#define TOT   512
#define NB    16
#define BD    32
#define HID   64
#define NLAY  4

// GEMM tiling: WG tile 128(M) x 256(N), K-step 32, 8 waves each owning 64x64
// Double-buffered LDS, one barrier per k-step.
#define TM 128
#define TN 256
#define TK 32
#define LDSS 40              // LDS row stride in halves (80B, 16B-aligned rows)
#define ABUF (TM * LDSS)     // one A buffer, in halves
#define BBUF (TN * LDSS)     // one B buffer, in halves

typedef __attribute__((ext_vector_type(8)))  float          v8f;
typedef __attribute__((ext_vector_type(16))) __bf16         v16bf;
typedef __attribute__((ext_vector_type(16))) unsigned short us16;
typedef __attribute__((ext_vector_type(8)))  unsigned short us8;
typedef __attribute__((ext_vector_type(4)))  unsigned short us4;

union BV { us16 u; v16bf v; };

// f32 -> bf16 round-to-nearest-even
__device__ __forceinline__ unsigned short f2bf(float f) {
    unsigned u = __float_as_uint(f);
    u += 0x7FFFu + ((u >> 16) & 1u);
    return (unsigned short)(u >> 16);
}
__device__ __forceinline__ float bf2f(unsigned short h) {
    return __uint_as_float(((unsigned)h) << 16);
}
__device__ __forceinline__ float geluf(float x) {
    return 0.5f * x * (1.0f + erff(x * 0.70710678118654752f));
}
__device__ __forceinline__ float softplusf(float x) {
    return (x > 20.0f) ? x : log1pf(__expf(x));
}

// ---------------------------------------------------------------------------
// Weight prep kernels
// ---------------------------------------------------------------------------
__global__ void cvt_bf16_kernel(const float* __restrict__ s,
                                unsigned short* __restrict__ d, int n) {
    int i = blockIdx.x * 256 + threadIdx.x;
    if (i < n) d[i] = f2bf(s[i]);
}

// mix_w [L,NB,NB,D,D] -> per-layer row-major [512][512]:  W[n][k]=mw[i,j,e,d]
// n = i*32+e, k = j*32+d
__global__ void mix_permute_kernel(const float* __restrict__ s,
                                   unsigned short* __restrict__ d) {
    int idx = blockIdx.x * 256 + threadIdx.x;          // 4*512*512 = 1048576
    int l = idx >> 18;
    int r = idx & 262143;
    int n = r >> 9;
    int k = r & 511;
    int i = n >> 5, e = n & 31;
    int j = k >> 5, dd = k & 31;
    size_t src = ((((size_t)l * 16 + i) * 16 + j) * 32 + e) * 32 + dd;
    d[idx] = f2bf(s[src]);
}

// ---------------------------------------------------------------------------
// Per-row norms -> tiny MLP -> softplus scales  (one thread per batch row)
// ---------------------------------------------------------------------------
__global__ __launch_bounds__(256)
void scales_kernel(const unsigned short* __restrict__ z,
                   const float* __restrict__ m1w, const float* __restrict__ m1b,
                   const float* __restrict__ m2w, const float* __restrict__ m2b,
                   const float* __restrict__ m3w, float* __restrict__ sOut) {
    __shared__ float sW1[HID * NB];
    __shared__ float sB1[HID];
    __shared__ float sW2[HID * HID];
    __shared__ float sB2[HID];
    __shared__ float sW3[NB * HID];
    int tid = threadIdx.x;
    for (int i = tid; i < HID * NB; i += 256) sW1[i] = m1w[i];
    for (int i = tid; i < HID * HID; i += 256) sW2[i] = m2w[i];
    for (int i = tid; i < NB * HID; i += 256) sW3[i] = m3w[i];
    if (tid < HID) { sB1[tid] = m1b[tid]; sB2[tid] = m2b[tid]; }
    __syncthreads();

    int row = blockIdx.x * 256 + tid;
    const unsigned short* zr = z + (size_t)row * TOT;

    float norms[NB];
#pragma unroll
    for (int i = 0; i < NB; ++i) {
        float acc = 0.0f;
#pragma unroll
        for (int c = 0; c < 4; ++c) {
            us8 u = *(const us8*)(zr + i * BD + c * 8);
#pragma unroll
            for (int e = 0; e < 8; ++e) { float v = bf2f(u[e]); acc += v * v; }
        }
        norms[i] = sqrtf(acc) + 1e-8f;
    }
    float h1[HID];
#pragma unroll 4
    for (int o = 0; o < HID; ++o) {
        float a = sB1[o];
#pragma unroll
        for (int i = 0; i < NB; ++i) a += norms[i] * sW1[o * NB + i];
        h1[o] = geluf(a);
    }
    float out[NB];
#pragma unroll
    for (int i = 0; i < NB; ++i) out[i] = 0.0f;
    for (int o = 0; o < HID; ++o) {
        float a = sB2[o];
#pragma unroll 8
        for (int i = 0; i < HID; ++i) a += h1[i] * sW2[o * HID + i];
        float h2 = geluf(a);
#pragma unroll
        for (int i = 0; i < NB; ++i) out[i] += sW3[i * HID + o] * h2;
    }
#pragma unroll
    for (int i = 0; i < NB; ++i) sOut[(size_t)row * NB + i] = softplusf(out[i]);
}

// ---------------------------------------------------------------------------
// WMMA GEMM:  Out[M,N] = epi( A[M,K] @ W[N,K]^T )
// ---------------------------------------------------------------------------
enum { EPI_BIAS_GELU = 0, EPI_BIAS = 1, EPI_BIAS_F32 = 2, EPI_MIX = 3 };

template <bool AF32, int EPI>
__global__ __launch_bounds__(256)
void gemm_wmma(const void* __restrict__ Aptr,
               const unsigned short* __restrict__ W,
               const float* __restrict__ bias,
               const unsigned short* __restrict__ zin,
               const float* __restrict__ scales,
               const float* __restrict__ gateb,
               void* __restrict__ Out, int M, int N, int K) {
    __shared__ unsigned short As[2 * ABUF];   // 2 x (128 x 32 (+pad))
    __shared__ unsigned short Bs[2 * BBUF];   // 2 x (256 x 32 (+pad))

    const int tid   = threadIdx.x;
    const int mBase = blockIdx.y * TM;
    const int nBase = blockIdx.x * TN;
    const int KT    = K / TK;                 // always even (16 or 32)

    // ---- global staging pointers (advance by TK per k-step) ----
    const int arow = tid >> 2, ak = (tid & 3) * 8;
    const unsigned short* A16 = (const unsigned short*)Aptr;
    const unsigned short* gA0 = A16 + (size_t)(mBase + arow) * K + ak;
    const unsigned short* gA1 = A16 + (size_t)(mBase + arow + 64) * K + ak;
    const float* Afp = (const float*)Aptr;
    const float* gAF[4];
    int afRow[4], afK[4];
#pragma unroll
    for (int j = 0; j < 4; ++j) {
        int c = tid + j * 256;
        afRow[j] = c >> 3; afK[j] = (c & 7) * 4;
        gAF[j] = Afp + (size_t)(mBase + afRow[j]) * K + afK[j];
    }
    const unsigned short* gB[4];
    int bRow[4], bK[4];
#pragma unroll
    for (int j = 0; j < 4; ++j) {
        int c = tid + j * 256;
        bRow[j] = c >> 2; bK[j] = (c & 3) * 8;
        gB[j] = W + (size_t)(nBase + bRow[j]) * K + bK[j];
    }

    // ---- LDS addresses for both buffers (invariant) ----
    unsigned short* sA0[2]; unsigned short* sA1[2];
    unsigned short* sAF[2][4]; unsigned short* sB[2][4];
    const unsigned short* fA[2][4]; const unsigned short* fB[2][4];

    const int lane  = tid & 31;
    const int wave  = tid >> 5;
    const int wm    = wave & 1;       // 2 M-groups of 64
    const int wn    = wave >> 1;      // 4 N-groups of 64
    const int laneM = lane & 15;
    const int hi    = lane >> 4;

#pragma unroll
    for (int bufi = 0; bufi < 2; ++bufi) {
        unsigned short* Ab = &As[bufi * ABUF];
        unsigned short* Bb = &Bs[bufi * BBUF];
        sA0[bufi] = &Ab[arow * LDSS + ak];
        sA1[bufi] = &Ab[(arow + 64) * LDSS + ak];
#pragma unroll
        for (int j = 0; j < 4; ++j) sAF[bufi][j] = &Ab[afRow[j] * LDSS + afK[j]];
#pragma unroll
        for (int j = 0; j < 4; ++j) sB[bufi][j] = &Bb[bRow[j] * LDSS + bK[j]];
#pragma unroll
        for (int mt = 0; mt < 4; ++mt)
            fA[bufi][mt] = &Ab[(wm * 64 + mt * 16 + laneM) * LDSS + hi * 8];
#pragma unroll
        for (int nt = 0; nt < 4; ++nt)
            fB[bufi][nt] = &Bb[(wn * 64 + nt * 16 + laneM) * LDSS + hi * 8];
    }

    // ---- staging registers ----
    us8    aH[2];
    float4 aF[4];
    us8    bH[4];

    v8f zero = {};
    v8f acc[4][4];
#pragma unroll
    for (int i = 0; i < 4; ++i)
#pragma unroll
        for (int j = 0; j < 4; ++j) acc[i][j] = zero;

    // helpers (always inlined with literal buffer index)
    auto gload = [&]() {
        if (AF32) {
#pragma unroll
            for (int j = 0; j < 4; ++j) { aF[j] = *(const float4*)gAF[j]; gAF[j] += TK; }
        } else {
            aH[0] = *(const us8*)gA0; gA0 += TK;
            aH[1] = *(const us8*)gA1; gA1 += TK;
        }
#pragma unroll
        for (int j = 0; j < 4; ++j) { bH[j] = *(const us8*)gB[j]; gB[j] += TK; }
    };
    auto sstore = [&](int bufi) {
        if (AF32) {
#pragma unroll
            for (int j = 0; j < 4; ++j) {
                us4 h;
                h[0] = f2bf(aF[j].x); h[1] = f2bf(aF[j].y);
                h[2] = f2bf(aF[j].z); h[3] = f2bf(aF[j].w);
                *(us4*)sAF[bufi][j] = h;
            }
        } else {
            *(us8*)sA0[bufi] = aH[0];
            *(us8*)sA1[bufi] = aH[1];
        }
#pragma unroll
        for (int j = 0; j < 4; ++j) *(us8*)sB[bufi][j] = bH[j];
    };
    auto compute = [&](int bufi) {
        BV a[4];
#pragma unroll
        for (int mt = 0; mt < 4; ++mt) {
            ((us8*)&a[mt].u)[0] = *(const us8*)fA[bufi][mt];
            ((us8*)&a[mt].u)[1] = *(const us8*)(fA[bufi][mt] + 16);
        }
        BV b[4];
#pragma unroll
        for (int nt = 0; nt < 4; ++nt) {
            ((us8*)&b[nt].u)[0] = *(const us8*)fB[bufi][nt];
            ((us8*)&b[nt].u)[1] = *(const us8*)(fB[bufi][nt] + 16);
        }
#pragma unroll
        for (int mt = 0; mt < 4; ++mt)
#pragma unroll
            for (int nt = 0; nt < 4; ++nt)
                acc[mt][nt] = __builtin_amdgcn_wmma_f32_16x16x32_bf16(
                    false, a[mt].v, false, b[nt].v, (short)0, acc[mt][nt],
                    false, false);
    };

    // ---- prologue: tile0 -> buf0, tile1 staged in regs ----
    gload();            // tile 0
    sstore(0);
    gload();            // tile 1
    __syncthreads();

    // ---- main loop: one barrier per k-step, ping-pong buffers ----
    for (int kt = 0; kt < KT; kt += 2) {
        const bool haveNext = (kt + 2 < KT);
        // even step: regs hold tile kt+1
        sstore(1);
        if (haveNext) gload();            // tile kt+2
        compute(0);                       // tile kt
        __syncthreads();
        // odd step: regs hold tile kt+2 (if any)
        if (haveNext) {
            sstore(0);
            gload();                      // tile kt+3 (KT even => exists)
        }
        compute(1);                       // tile kt+1
        __syncthreads();
    }

    // ---- epilogue ----
#pragma unroll
    for (int mt = 0; mt < 4; ++mt) {
#pragma unroll
        for (int nt = 0; nt < 4; ++nt) {
            int rBase = mBase + wm * 64 + mt * 16 + hi * 8;
            int c     = nBase + wn * 64 + nt * 16 + laneM;
            float bc  = (EPI == EPI_MIX) ? 0.0f : bias[c];
            float g   = 0.0f;
            if (EPI == EPI_MIX) g = 1.0f / (1.0f + __expf(-gateb[c >> 5]));
#pragma unroll
            for (int v = 0; v < 8; ++v) {
                int   r   = rBase + v;
                float val = acc[mt][nt][v];
                if (EPI == EPI_BIAS_GELU) {
                    val = geluf(val + bc);
                    ((unsigned short*)Out)[(size_t)r * N + c] = f2bf(val);
                } else if (EPI == EPI_BIAS) {
                    ((unsigned short*)Out)[(size_t)r * N + c] = f2bf(val + bc);
                } else if (EPI == EPI_BIAS_F32) {
                    ((float*)Out)[(size_t)r * N + c] = val + bc;
                } else { // EPI_MIX: z*(1+softplus_scale) + sigmoid(gate)*mixed
                    float zf = bf2f(zin[(size_t)r * N + c]);
                    float sc = scales[(size_t)r * NB + (c >> 5)];
                    val = zf * (1.0f + sc) + g * val;
                    ((unsigned short*)Out)[(size_t)r * N + c] = f2bf(val);
                }
            }
        }
    }
}

// ---------------------------------------------------------------------------
// Host launch
// ---------------------------------------------------------------------------
extern "C" void kernel_launch(void* const* d_in, const int* in_sizes, int n_in,
                              void* d_out, int out_size, void* d_ws, size_t ws_size,
                              hipStream_t stream) {
    (void)in_sizes; (void)n_in; (void)out_size; (void)ws_size;
    const float* x    = (const float*)d_in[0];
    const float* ew1  = (const float*)d_in[1];
    const float* eb1  = (const float*)d_in[2];
    const float* ew2  = (const float*)d_in[3];
    const float* eb2  = (const float*)d_in[4];
    const float* ew3  = (const float*)d_in[5];
    const float* eb3  = (const float*)d_in[6];
    const float* m1w  = (const float*)d_in[7];
    const float* m1b  = (const float*)d_in[8];
    const float* m2w  = (const float*)d_in[9];
    const float* m2b  = (const float*)d_in[10];
    const float* m3w  = (const float*)d_in[11];
    const float* mixw = (const float*)d_in[12];
    const float* gb   = (const float*)d_in[13];
    const float* dw1  = (const float*)d_in[14];
    const float* db1  = (const float*)d_in[15];
    const float* dw2  = (const float*)d_in[16];
    const float* db2  = (const float*)d_in[17];
    const float* dw3  = (const float*)d_in[18];
    const float* db3  = (const float*)d_in[19];

    unsigned char* ws = (unsigned char*)d_ws;
    size_t off = 0;
    auto alloc = [&](size_t bytes) {
        void* p = ws + off;
        off += (bytes + 255) & ~(size_t)255;
        return p;
    };
    unsigned short* wEnc1 = (unsigned short*)alloc((size_t)TOT * DIN * 2);
    unsigned short* wEnc2 = (unsigned short*)alloc((size_t)TOT * TOT * 2);
    unsigned short* wEnc3 = (unsigned short*)alloc((size_t)TOT * TOT * 2);
    unsigned short* wDec1 = (unsigned short*)alloc((size_t)TOT * TOT * 2);
    unsigned short* wDec2 = (unsigned short*)alloc((size_t)TOT * TOT * 2);
    unsigned short* wDec3 = (unsigned short*)alloc((size_t)DOUT * TOT * 2);
    unsigned short* wMix  = (unsigned short*)alloc((size_t)NLAY * TOT * TOT * 2);
    unsigned short* zA    = (unsigned short*)alloc((size_t)BATCH * TOT * 2);
    unsigned short* zB    = (unsigned short*)alloc((size_t)BATCH * TOT * 2);
    float*          sBuf  = (float*)alloc((size_t)BATCH * NB * 4);

    auto cvt = [&](const float* s, unsigned short* d, int n) {
        cvt_bf16_kernel<<<(n + 255) / 256, 256, 0, stream>>>(s, d, n);
    };
    cvt(ew1, wEnc1, TOT * DIN);
    cvt(ew2, wEnc2, TOT * TOT);
    cvt(ew3, wEnc3, TOT * TOT);
    cvt(dw1, wDec1, TOT * TOT);
    cvt(dw2, wDec2, TOT * TOT);
    cvt(dw3, wDec3, DOUT * TOT);
    mix_permute_kernel<<<(NLAY * TOT * TOT) / 256, 256, 0, stream>>>(mixw, wMix);

    dim3 blk(256);
    dim3 g512(TOT / TN, BATCH / TM);     // (2, 512)
    dim3 g1024(DOUT / TN, BATCH / TM);   // (4, 512)

    // encoder
    gemm_wmma<true, EPI_BIAS_GELU><<<g512, blk, 0, stream>>>(
        x, wEnc1, eb1, nullptr, nullptr, nullptr, zA, BATCH, TOT, DIN);
    gemm_wmma<false, EPI_BIAS_GELU><<<g512, blk, 0, stream>>>(
        zA, wEnc2, eb2, nullptr, nullptr, nullptr, zB, BATCH, TOT, TOT);
    gemm_wmma<false, EPI_BIAS><<<g512, blk, 0, stream>>>(
        zB, wEnc3, eb3, nullptr, nullptr, nullptr, zA, BATCH, TOT, TOT);

    // latent soft-equivariant layers
    for (int l = 0; l < NLAY; ++l) {
        const unsigned short* zc = (l & 1) ? zB : zA;
        unsigned short*       zn = (l & 1) ? zA : zB;
        scales_kernel<<<BATCH / 256, blk, 0, stream>>>(
            zc, m1w + l * HID * NB, m1b + l * HID, m2w + l * HID * HID,
            m2b + l * HID, m3w + l * NB * HID, sBuf);
        gemm_wmma<false, EPI_MIX><<<g512, blk, 0, stream>>>(
            zc, wMix + (size_t)l * TOT * TOT, nullptr, zc, sBuf, gb + l * NB,
            zn, BATCH, TOT, TOT);
    }
    // after 4 layers result is in zA
    gemm_wmma<false, EPI_BIAS_GELU><<<g512, blk, 0, stream>>>(
        zA, wDec1, db1, nullptr, nullptr, nullptr, zB, BATCH, TOT, TOT);
    gemm_wmma<false, EPI_BIAS_GELU><<<g512, blk, 0, stream>>>(
        zB, wDec2, db2, nullptr, nullptr, nullptr, zA, BATCH, TOT, TOT);
    gemm_wmma<false, EPI_BIAS_F32><<<g1024, blk, 0, stream>>>(
        zA, wDec3, db3, nullptr, nullptr, nullptr, d_out, BATCH, DOUT, TOT);
}
